// MyCompGCN_70815420776924
// MI455X (gfx1250) — compile-verified, matrix operand order checked by reading, not run on hardware
//
#include <hip/hip_runtime.h>
#include <hip/hip_bf16.h>

// ---------------------------------------------------------------------------
// CompGCN layer for MI455X (gfx1250, wave32, WMMA).
// h[dst] += (node_feat[src]*edge_feat[r]*norm) @ W_R[r]   via f16 WMMA, f32 acc
// then BatchNorm(train, biased var) + tanh;  plus edge_feat @ W_rel.
// Working set (node_feat 25.6MB, h 25.6MB, W 2MB) fits in 192MB L2 -> compute
// bound; the 26.2 GFLOP edge-matmul goes through V_WMMA_F32_16X16X32_F16.
// Scatter is branchless: padding rows add exact zeros into a dummy h row.
// ---------------------------------------------------------------------------

#define CN   50000
#define CE   800000
#define CNR  32
#define CD   128
#define CEPS 1e-5f

typedef _Float16 v16h __attribute__((ext_vector_type(16)));
typedef _Float16 v8h  __attribute__((ext_vector_type(8)));
typedef _Float16 v4h  __attribute__((ext_vector_type(4)));
typedef float    v8f  __attribute__((ext_vector_type(8)));

// -------------------- prep: transpose+convert W_R -> Wt[r][n][k] (f16) ------
__global__ void prep_wt_kernel(const float* __restrict__ W_R,
                               _Float16* __restrict__ Wt) {
    int i = blockIdx.x * blockDim.x + threadIdx.x;           // NR*128*128
    if (i >= CNR * CD * CD) return;
    int r   = i >> 14;
    int rem = i & 16383;
    int n   = rem >> 7;
    int k   = rem & 127;
    Wt[i] = (_Float16)W_R[(r << 14) + (k << 7) + n];
}

// -------------------- bucket edges by relation ------------------------------
__global__ void count_kernel(const int* __restrict__ etype,
                             int* __restrict__ counts) {
    int e = blockIdx.x * blockDim.x + threadIdx.x;
    if (e < CE) atomicAdd(&counts[etype[e]], 1);
}

__global__ void scan_kernel(const int* __restrict__ counts,
                            int* __restrict__ offsets,
                            int* __restrict__ cursors) {
    if (threadIdx.x == 0) {
        int acc = 0;
        for (int r = 0; r < CNR; ++r) {
            offsets[r] = acc;
            cursors[r] = acc;
            acc += (counts[r] + 15) & ~15;   // pad each bucket to 16 edges
        }
        offsets[CNR] = acc;
    }
}

__global__ void bucket_kernel(const int* __restrict__ etype,
                              int* __restrict__ cursors,
                              int* __restrict__ edge_ids) {
    int e = blockIdx.x * blockDim.x + threadIdx.x;
    if (e < CE) {
        int pos = atomicAdd(&cursors[etype[e]], 1);
        edge_ids[pos] = e;
    }
}

// -------------------- main WMMA gemm + scatter ------------------------------
// 1 wave (32 lanes) per tile of 16 same-relation edges. 8 waves / block.
// h has CN+1 rows: row CN is a dummy sink for padding rows (they add 0.0).
__global__ __launch_bounds__(256)
void gemm_scatter_kernel(const float* __restrict__ node_feat,
                         const float* __restrict__ edge_feat,
                         const _Float16* __restrict__ Wt,
                         const float* __restrict__ etype_norm,
                         const int* __restrict__ src,
                         const int* __restrict__ dst,
                         const int* __restrict__ edge_ids,
                         const int* __restrict__ offsets,
                         float* __restrict__ h) {
    __shared__ int soff[CNR + 1];
    __shared__ _Float16 lds[8][16 * CD];     // 4KB per wave

    if (threadIdx.x <= CNR) soff[threadIdx.x] = offsets[threadIdx.x];
    __syncthreads();

    const int wave  = threadIdx.x >> 5;
    const int lane  = threadIdx.x & 31;
    const int t     = blockIdx.x * 8 + wave;
    const int t16   = t * 16;
    const int total = soff[CNR];
    const bool active = (t16 < total);       // wave-uniform

    // relation of this tile (buckets are 16-aligned -> tile never straddles)
    int r = 0;
    if (active) {
        int lo = 0, hi = CNR;
        while (hi - lo > 1) {
            int mid = (lo + hi) >> 1;
            if (soff[mid] <= t16) lo = mid; else hi = mid;
        }
        r = lo;
    }

    // lanes 0..15 own one edge slot each; padding => src=0, norm=0, dst=CN
    int dstv = CN, srcv = 0;
    float normv = 0.f;
    if (active && lane < 16) {
        int eid = edge_ids[t16 + lane];
        if (eid >= 0) {
            srcv  = src[eid];
            dstv  = dst[eid];
            normv = etype_norm[eid];
        }
    }

    _Float16* myLds = lds[wave];

    // ---- compose 16 rows of [E-tile, 128] in f32, round to f16 into LDS ----
    // Branchless: padding rows load node_feat[0] and multiply by norm=0.
    if (active) {
        const float4 ef = *(const float4*)(edge_feat + r * CD + lane * 4);
        #pragma unroll
        for (int m = 0; m < 16; ++m) {
            int   sm = __shfl(srcv, m);
            float nm = __shfl(normv, m);
            const float4 nf =
                *(const float4*)(node_feat + (size_t)sm * CD + lane * 4);
            v4h out;
            out[0] = (_Float16)(nf.x * ef.x * nm);
            out[1] = (_Float16)(nf.y * ef.y * nm);
            out[2] = (_Float16)(nf.z * ef.z * nm);
            out[3] = (_Float16)(nf.w * ef.w * nm);
            *(v4h*)(myLds + m * CD + lane * 4) = out;   // ds_store_b64
        }
    }
    __syncthreads();

    if (active) {
        const int m0    = lane & 15;   // A: both lane halves map M=0..15
        const int khalf = lane >> 4;   // selects K group per ISA 7.12.2

        // A fragments: 4 chunks of K=32; each = two contiguous ds_load_b128
        v16h a[4];
        #pragma unroll
        for (int c = 0; c < 4; ++c) {
            v8h alo = *(const v8h*)(myLds + m0 * CD + c * 32 + khalf * 8);
            v8h ahi = *(const v8h*)(myLds + m0 * CD + c * 32 + 16 + khalf * 8);
            a[c] = __builtin_shufflevector(alo, ahi,
                     0,1,2,3,4,5,6,7,8,9,10,11,12,13,14,15);
        }

        // D layout: lane half holds rows M = khalf*8 + v in acc element v.
        // Precompute 8 destination row pointers once; j-loop offsets are
        // compile-time constants that fold into the atomic's immediate field.
        float* rowp[8];
        #pragma unroll
        for (int v = 0; v < 8; ++v) {
            int dv = __shfl(dstv, (khalf << 3) + v);    // CN for padding
            rowp[v] = h + (size_t)dv * CD + m0;
        }

        const _Float16* wbase = Wt + (size_t)r * CD * CD;  // [n][k], f16

        #pragma unroll
        for (int j = 0; j < 8; ++j) {                      // 8 output N-tiles
            v8f acc = {};
            const _Float16* wrow = wbase + (size_t)(j * 16 + m0) * CD;
            #pragma unroll
            for (int c = 0; c < 4; ++c) {                  // K = 128 / 32
                v8h blo = *(const v8h*)(wrow + c * 32 + khalf * 8);
                v8h bhi = *(const v8h*)(wrow + c * 32 + 16 + khalf * 8);
                v16h b = __builtin_shufflevector(blo, bhi,
                           0,1,2,3,4,5,6,7,8,9,10,11,12,13,14,15);
                acc = __builtin_amdgcn_wmma_f32_16x16x32_f16(
                          false, a[c], false, b, (short)0, acc, false, false);
            }
            #pragma unroll
            for (int v = 0; v < 8; ++v)
                atomicAdd(rowp[v] + j * 16, acc[v]);       // branchless scatter
        }
    }
}

// -------------------- BatchNorm stats (col sum / sumsq over N rows) ---------
__global__ __launch_bounds__(128)
void bn_stats_kernel(const float* __restrict__ h, float* __restrict__ stats) {
    int c = threadIdx.x;
    float s = 0.f, s2 = 0.f;
    for (int n = blockIdx.x; n < CN; n += gridDim.x) {
        float x = h[(size_t)n * CD + c];
        s += x;
        s2 += x * x;
    }
    atomicAdd(&stats[c], s);
    atomicAdd(&stats[CD + c], s2);
}

// -------------------- BN apply + tanh ---------------------------------------
__global__ void bn_apply_kernel(const float* __restrict__ h,
                                const float* __restrict__ stats,
                                const float* __restrict__ gamma,
                                const float* __restrict__ beta,
                                float* __restrict__ out) {
    size_t i = (size_t)blockIdx.x * blockDim.x + threadIdx.x;
    if (i >= (size_t)CN * CD) return;
    int c = (int)(i & (CD - 1));
    const float invN = 1.0f / (float)CN;
    float mean = stats[c] * invN;
    float var  = stats[CD + c] * invN - mean * mean;   // biased variance
    float y = (h[i] - mean) * rsqrtf(var + CEPS) * gamma[c] + beta[c];
    out[i] = tanhf(y);
}

// -------------------- tiny side output: edge_feat @ W_rel -------------------
__global__ void wrel_kernel(const float* __restrict__ edge_feat,
                            const float* __restrict__ W_rel,
                            float* __restrict__ out2) {
    int i = blockIdx.x * blockDim.x + threadIdx.x;     // NR*D
    if (i >= CNR * CD) return;
    int r = i >> 7, o = i & 127;
    float s = 0.f;
    #pragma unroll 8
    for (int k = 0; k < CD; ++k)
        s += edge_feat[r * CD + k] * W_rel[k * CD + o];
    out2[i] = s;
}

// ---------------------------------------------------------------------------
extern "C" void kernel_launch(void* const* d_in, const int* in_sizes, int n_in,
                              void* d_out, int out_size, void* d_ws, size_t ws_size,
                              hipStream_t stream) {
    const float* node_feat = (const float*)d_in[0];
    const float* edge_feat = (const float*)d_in[1];
    const float* W_R       = (const float*)d_in[2];
    const float* W_rel     = (const float*)d_in[3];
    const float* bn_gamma  = (const float*)d_in[4];
    const float* bn_beta   = (const float*)d_in[5];
    const float* enorm     = (const float*)d_in[6];
    const int*   src       = (const int*)d_in[7];
    const int*   dst       = (const int*)d_in[8];
    const int*   etype     = (const int*)d_in[9];
    float* out = (float*)d_out;

    // ---- workspace layout (256B aligned slabs) ----
    char* ws = (char*)d_ws;
    float* h       = (float*)ws;                       // (N+1)*D f32 accum
    size_t off = (size_t)(CN + 1) * CD * 4;            // 25,600,512 (256-mult)
    int*   counts  = (int*)(ws + off);  off += 256;    // 32 ints (padded)
    int*   cursors = (int*)(ws + off);  off += 256;    // 32 ints (padded)
    float* stats   = (float*)(ws + off); off += 1024;  // sum[128], sumsq[128]
    size_t zero_bytes = off;                           // h..stats zeroed
    int*   offsets = (int*)(ws + off);  off += 256;    // 33 ints (padded)
    _Float16* Wt   = (_Float16*)(ws + off);
    off += (size_t)CNR * CD * CD * 2;                  // 1 MB
    int* edge_ids  = (int*)(ws + off);
    const size_t eid_count = (size_t)CE + 16 * CNR;    // 16-padded buckets
    const size_t eid_bytes = eid_count * 4;

    hipMemsetAsync(ws, 0, zero_bytes, stream);
    hipMemsetAsync(edge_ids, 0xFF, eid_bytes, stream); // fill with -1

    prep_wt_kernel<<<(CNR * CD * CD + 255) / 256, 256, 0, stream>>>(W_R, Wt);
    count_kernel <<<(CE + 255) / 256, 256, 0, stream>>>(etype, counts);
    scan_kernel  <<<1, 32, 0, stream>>>(counts, offsets, cursors);
    bucket_kernel<<<(CE + 255) / 256, 256, 0, stream>>>(etype, cursors, edge_ids);

    const int tiles  = (int)(eid_count / 16);          // 50032
    const int blocks = (tiles + 7) / 8;                // 8 waves per block
    gemm_scatter_kernel<<<blocks, 256, 0, stream>>>(
        node_feat, edge_feat, Wt, enorm, src, dst, edge_ids, offsets, h);

    bn_stats_kernel<<<256, 128, 0, stream>>>(h, stats);
    bn_apply_kernel<<<((size_t)CN * CD + 255) / 256, 256, 0, stream>>>(
        h, stats, bn_gamma, bn_beta, out);
    wrel_kernel<<<(CNR * CD + 255) / 256, 256, 0, stream>>>(
        edge_feat, W_rel, out + (size_t)CN * CD);
}
